// GradMimicryNetwork_23227183137547
// MI455X (gfx1250) — matrix operand order repaired and checked by reading, not compiled, not run on hardware
//
#include <hip/hip_runtime.h>

// ---------------------------------------------------------------------------
// SNN forward restructured as 5 layer-wise GEMMs (M = T*B = 8192) + LIF scans.
// GEMMs use v_wmma_f32_16x16x32_f16 with double-buffered LDS tiles filled by
// gfx1250 async loads (global_load_async_to_lds_b128, ASYNCcnt-tracked).
// ---------------------------------------------------------------------------

typedef __attribute__((ext_vector_type(16))) _Float16 v16h;
typedef __attribute__((ext_vector_type(8)))  float    v8f;

#define TSTEPS 128
#define BATCH  64
#define MROWS  (TSTEPS * BATCH)   // 8192
#define BK     64                 // K-tile per stage
#define PITCH  72                 // 64 halves + 8 pad (144B rows, 16B aligned)
#define TILEHALVES (128 * PITCH)  // halves per (A or B) tile

// ---------------------------------------------------------------------------
// Weight prep: relu(w) -> f16, stored transposed as Wt[N][K]
// ---------------------------------------------------------------------------
__global__ void prep_w_kernel(const float* __restrict__ w, _Float16* __restrict__ Wt,
                              int K, int N) {
    int idx = blockIdx.x * 256 + threadIdx.x;   // idx = n*K + k
    int n = idx / K;
    int k = idx - n * K;
    float x = w[(size_t)k * N + n];
    Wt[idx] = (_Float16)(x > 0.0f ? x : 0.0f);
}

// ---------------------------------------------------------------------------
// Input transpose: in[B][1024][T] f32 -> A0[(t*B+b)][1024] f16
// ---------------------------------------------------------------------------
__global__ void prep_input_kernel(const float* __restrict__ in, _Float16* __restrict__ A0) {
    int idx = blockIdx.x * 256 + threadIdx.x;   // (t*B + b)*1024 + k
    int t   = idx >> 16;
    int rem = idx & 0xFFFF;
    int b   = rem >> 10;
    int k   = rem & 1023;
    A0[idx] = (_Float16)in[(b << 17) + (k << 7) + t];
}

// ---------------------------------------------------------------------------
// GEMM: C[M,N] (f32) = A[M,K] (f16 row-major) * W (f16, stored as Wt[N][K])
// 256 threads = 8 waves; 128x128 block tile; BK=64; double-buffered LDS
// filled via async loads; each wave computes a 32x64 region (2x4 WMMA tiles).
// ---------------------------------------------------------------------------
__global__ __launch_bounds__(256) void gemm_f16_kernel(
    const _Float16* __restrict__ A, const _Float16* __restrict__ Wt,
    float* __restrict__ C, int K, int N) {

    __shared__ _Float16 lds[2][2 * TILEHALVES];   // [buf][ A tile | B tile ]

    const int tid   = threadIdx.x;
    const int lane  = tid & 31;
    const int wave  = tid >> 5;
    const int waveM = wave & 3;      // 4 waves along M
    const int waveN = wave >> 2;     // 2 waves along N
    const int lhi   = lane >> 4;     // lane half-group (0/1)
    const int llo   = lane & 15;
    const int mBase = blockIdx.y * 128;
    const int nBase = blockIdx.x * 128;

    // 32-bit LDS byte address of the shared block (flat addr low bits = LDS offset)
    const unsigned ldsBase = (unsigned)(unsigned long long)&lds[0][0];

    v8f acc[2][4] = {};

    // Issue one stage of async global->LDS copies (8 x B128 per thread).
    auto stage = [&](int k0, int buf) {
        const unsigned aBase = ldsBase + (unsigned)buf * (2u * TILEHALVES * 2u);
        const unsigned bBase = aBase + (unsigned)TILEHALVES * 2u;
#pragma unroll
        for (int i = 0; i < 4; ++i) {
            int q  = tid + i * 256;          // 0..1023
            int r  = q >> 3;                 // row 0..127
            int c8 = (q & 7) << 3;           // half col 0,8,...,56
            const _Float16* ga = &A[(size_t)(mBase + r) * K + k0 + c8];
            unsigned la = aBase + (unsigned)(r * PITCH + c8) * 2u;
            asm volatile("global_load_async_to_lds_b128 %0, %1, off"
                         :: "v"(la), "v"(ga) : "memory");
            const _Float16* gb = &Wt[(size_t)(nBase + r) * K + k0 + c8];
            unsigned lb = bBase + (unsigned)(r * PITCH + c8) * 2u;
            asm volatile("global_load_async_to_lds_b128 %0, %1, off"
                         :: "v"(lb), "v"(gb) : "memory");
        }
    };

    const int nk = K / BK;
    stage(0, 0);

    for (int ki = 0; ki < nk; ++ki) {
        const int cur = ki & 1;
        if (ki + 1 < nk) {
            stage((ki + 1) * BK, cur ^ 1);
            // 8 newer loads outstanding -> older stage (ours) has completed
            asm volatile("s_wait_asynccnt 8" ::: "memory");
        } else {
            asm volatile("s_wait_asynccnt 0" ::: "memory");
        }
        __syncthreads();   // all waves' async fills of buf[cur] visible

        const _Float16* tA = &lds[cur][0];
        const _Float16* tB = &lds[cur][TILEHALVES];

#pragma unroll
        for (int kt = 0; kt < 2; ++kt) {
            union Frag { v16h v; uint4 u[2]; };
            Frag af[2], bf[4];
#pragma unroll
            for (int mt = 0; mt < 2; ++mt) {
                // A: lane holds row M=llo; K chunks {8*lhi+0..7, 8*lhi+16..23}
                int row = waveM * 32 + mt * 16 + llo;
                int h0  = row * PITCH + kt * 32 + lhi * 8;
                af[mt].u[0] = *(const uint4*)(&tA[h0]);
                af[mt].u[1] = *(const uint4*)(&tA[h0 + 16]);
            }
#pragma unroll
            for (int nt = 0; nt < 4; ++nt) {
                // B: lane holds column N=llo; K chunks {8*lhi+0..7, 8*lhi+16..23}
                int row = waveN * 64 + nt * 16 + llo;
                int h0  = row * PITCH + kt * 32 + lhi * 8;
                bf[nt].u[0] = *(const uint4*)(&tB[h0]);
                bf[nt].u[1] = *(const uint4*)(&tB[h0 + 16]);
            }
#pragma unroll
            for (int mt = 0; mt < 2; ++mt)
#pragma unroll
                for (int nt = 0; nt < 4; ++nt)
                    acc[mt][nt] = __builtin_amdgcn_wmma_f32_16x16x32_f16(
                        false, af[mt].v, false, bf[nt].v,
                        (short)0, acc[mt][nt], false, false);
        }
        __syncthreads();   // reads of buf[cur] done -> safe to refill next iter
    }

    // ---- store D: VGPR r -> row (r + 8*lhi), col = llo ----
#pragma unroll
    for (int mt = 0; mt < 2; ++mt) {
#pragma unroll
        for (int nt = 0; nt < 4; ++nt) {
            int col   = nBase + waveN * 64 + nt * 16 + llo;
            int rbase = mBase + waveM * 32 + mt * 16 + lhi * 8;
#pragma unroll
            for (int r = 0; r < 8; ++r)
                C[(size_t)(rbase + r) * N + col] = acc[mt][nt][r];
        }
    }
}

// ---------------------------------------------------------------------------
// LIF dynamics: v = 0.9*v + I; s = (v>=1); v = s ? 0 : v
// ---------------------------------------------------------------------------
__device__ __forceinline__ float lif_step(float& v, float cur) {
    v = 0.9f * v + cur;
    float s = (v >= 1.0f) ? 1.0f : 0.0f;
    v -= s * v;
    return s;
}

// Hidden-layer scan: I[(t*B+b)*H + h] -> spikes_out[B][H][T] (f32) and
// Snext[(t+1)*B+b][H] (f16, time-shifted GEMM operand; row block 0 zeroed).
__global__ void lif_hidden_kernel(const float* __restrict__ I, int H,
                                  float* __restrict__ spikes_out,
                                  _Float16* __restrict__ Snext) {
    int idx = blockIdx.x * 256 + threadIdx.x;   // b*H + h
    const int stride = BATCH * H;
    Snext[idx] = (_Float16)0.0f;                // s_prev at t=0 is zero
    float v = 0.0f;
    float* outp = spikes_out + (size_t)idx * TSTEPS;
    for (int t = 0; t < TSTEPS; t += 4) {
        float4 o;
        o.x = lif_step(v, I[(size_t)(t + 0) * stride + idx]);
        o.y = lif_step(v, I[(size_t)(t + 1) * stride + idx]);
        o.z = lif_step(v, I[(size_t)(t + 2) * stride + idx]);
        o.w = lif_step(v, I[(size_t)(t + 3) * stride + idx]);
        *(float4*)(outp + t) = o;
        Snext[(size_t)(t + 1) * stride + idx] = (_Float16)o.x;
        Snext[(size_t)(t + 2) * stride + idx] = (_Float16)o.y;
        Snext[(size_t)(t + 3) * stride + idx] = (_Float16)o.z;
        if (t + 4 < TSTEPS)
            Snext[(size_t)(t + 4) * stride + idx] = (_Float16)o.w;
    }
}

// Output-layer scan: also produces firing rates (mean over t).
__global__ void lif_out_kernel(const float* __restrict__ I,
                               float* __restrict__ spikes_out,
                               float* __restrict__ rates) {
    int idx = blockIdx.x * 256 + threadIdx.x;   // b*128 + h
    const int stride = BATCH * 128;
    float v = 0.0f, sum = 0.0f;
    float* outp = spikes_out + (size_t)idx * TSTEPS;
    for (int t = 0; t < TSTEPS; t += 4) {
        float4 o;
        o.x = lif_step(v, I[(size_t)(t + 0) * stride + idx]);
        o.y = lif_step(v, I[(size_t)(t + 1) * stride + idx]);
        o.z = lif_step(v, I[(size_t)(t + 2) * stride + idx]);
        o.w = lif_step(v, I[(size_t)(t + 3) * stride + idx]);
        *(float4*)(outp + t) = o;
        sum += o.x + o.y + o.z + o.w;
    }
    rates[idx] = sum * (1.0f / (float)TSTEPS);
}

// ---------------------------------------------------------------------------
extern "C" void kernel_launch(void* const* d_in, const int* in_sizes, int n_in,
                              void* d_out, int out_size, void* d_ws, size_t ws_size,
                              hipStream_t stream) {
    const float* in_spk = (const float*)d_in[0];            // [64,1024,128]
    const float* w0 = (const float*)d_in[1];                // [1024,2048]
    const float* w1 = (const float*)d_in[2];                // [2048,2048]
    const float* w2 = (const float*)d_in[3];                // [2048,1024]
    const float* w3 = (const float*)d_in[4];                // [1024,512]
    const float* w4 = (const float*)d_in[5];                // [512,128]

    // ---- scratch layout ----
    char* ws = (char*)d_ws;
    size_t off = 0;
    auto carve = [&](size_t bytes) {
        char* p = ws + off;
        off += (bytes + 255) & ~(size_t)255;
        return p;
    };
    _Float16* Wt0 = (_Float16*)carve((size_t)1024 * 2048 * 2);
    _Float16* Wt1 = (_Float16*)carve((size_t)2048 * 2048 * 2);
    _Float16* Wt2 = (_Float16*)carve((size_t)2048 * 1024 * 2);
    _Float16* Wt3 = (_Float16*)carve((size_t)1024 * 512 * 2);
    _Float16* Wt4 = (_Float16*)carve((size_t)512 * 128 * 2);
    _Float16* A0  = (_Float16*)carve((size_t)MROWS * 1024 * 2);
    _Float16* S0  = (_Float16*)carve((size_t)MROWS * 2048 * 2);
    _Float16* S1  = (_Float16*)carve((size_t)MROWS * 2048 * 2);
    float*    Icur = (float*)carve((size_t)MROWS * 2048 * 4);
    _Float16* S2 = S0;   // ping-pong reuse (S0 dead after GEMM1)
    _Float16* S3 = S1;   // (S1 dead after GEMM2)

    // ---- d_out layout (flat, reference return order) ----
    float* out = (float*)d_out;
    float* h0  = out;                                   // 64*2048*128
    float* h1  = h0 + (size_t)64 * 2048 * 128;
    float* h2  = h1 + (size_t)64 * 2048 * 128;          // 64*1024*128
    float* h3  = h2 + (size_t)64 * 1024 * 128;          // 64*512*128
    float* os  = h3 + (size_t)64 * 512 * 128;           // 64*128*128
    float* fr  = os + (size_t)64 * 128 * 128;           // 64*128

    // ---- prep: relu + f16 + transpose weights; transpose input ----
    prep_w_kernel<<<(1024 * 2048) / 256, 256, 0, stream>>>(w0, Wt0, 1024, 2048);
    prep_w_kernel<<<(2048 * 2048) / 256, 256, 0, stream>>>(w1, Wt1, 2048, 2048);
    prep_w_kernel<<<(2048 * 1024) / 256, 256, 0, stream>>>(w2, Wt2, 2048, 1024);
    prep_w_kernel<<<(1024 * 512) / 256, 256, 0, stream>>>(w3, Wt3, 1024, 512);
    prep_w_kernel<<<(512 * 128) / 256, 256, 0, stream>>>(w4, Wt4, 512, 128);
    prep_input_kernel<<<(MROWS * 1024) / 256, 256, 0, stream>>>(in_spk, A0);

    // ---- layer 0 ----
    gemm_f16_kernel<<<dim3(2048 / 128, MROWS / 128), 256, 0, stream>>>(A0, Wt0, Icur, 1024, 2048);
    lif_hidden_kernel<<<(BATCH * 2048) / 256, 256, 0, stream>>>(Icur, 2048, h0, S0);

    // ---- layer 1 ----
    gemm_f16_kernel<<<dim3(2048 / 128, MROWS / 128), 256, 0, stream>>>(S0, Wt1, Icur, 2048, 2048);
    lif_hidden_kernel<<<(BATCH * 2048) / 256, 256, 0, stream>>>(Icur, 2048, h1, S1);

    // ---- layer 2 ----
    gemm_f16_kernel<<<dim3(1024 / 128, MROWS / 128), 256, 0, stream>>>(S1, Wt2, Icur, 2048, 1024);
    lif_hidden_kernel<<<(BATCH * 1024) / 256, 256, 0, stream>>>(Icur, 1024, h2, S2);

    // ---- layer 3 ----
    gemm_f16_kernel<<<dim3(512 / 128, MROWS / 128), 256, 0, stream>>>(S2, Wt3, Icur, 1024, 512);
    lif_hidden_kernel<<<(BATCH * 512) / 256, 256, 0, stream>>>(Icur, 512, h3, S3);

    // ---- output layer ----
    gemm_f16_kernel<<<dim3(128 / 128, MROWS / 128), 256, 0, stream>>>(S3, Wt4, Icur, 512, 128);
    lif_out_kernel<<<(BATCH * 128) / 256, 256, 0, stream>>>(Icur, os, fr);
}
